// SingleStreamBlock_807453852411
// MI455X (gfx1250) — compile-verified, matrix operand order checked by reading, not run on hardware
//
#include <hip/hip_runtime.h>
#include <hip/hip_bf16.h>
#include <math.h>

// ---------------- problem sizes ----------------
#define HIDN   3072
#define NHEADS 24
#define HD     128
#define MLPD   12288
#define BBATCH 2
#define LSEQ   2048
#define M_TOT  (BBATCH*LSEQ)       // 4096 tokens
#define QKV_N  (3*HIDN)            // 9216
#define N1     (3*HIDN + MLPD)     // 21504  (lin1 out)
#define K1     HIDN                // 3072
#define N2     HIDN                // 3072   (lin2 out)
#define K2     (HIDN + MLPD)       // 15360  (lin2 in)

typedef unsigned int u32;
typedef __attribute__((ext_vector_type(4)))  u32    u32x4;
typedef __attribute__((ext_vector_type(16))) __bf16 v16bf;
typedef __attribute__((ext_vector_type(8)))  float  v8f;
typedef int v4i __attribute__((__vector_size__(16)));

// ---- CDNA5 async global->LDS path (guarded; falls back to sync staging) ----
#if defined(__has_builtin)
#if __has_builtin(__builtin_amdgcn_global_load_async_to_lds_b128) && \
    __has_builtin(__builtin_amdgcn_s_wait_asynccnt)
#define USE_ASYNC_LDS 1
#endif
#endif

#ifdef USE_ASYNC_LDS
__device__ __forceinline__ void async_cp16(const __bf16* g, __bf16* l) {
  __builtin_amdgcn_global_load_async_to_lds_b128(
      (__attribute__((address_space(1))) v4i*)g,
      (__attribute__((address_space(3))) v4i*)l,
      0, 0);
}
#endif

// ---------------- helpers ----------------
__device__ __forceinline__ float wave_sum(float v) {
  #pragma unroll
  for (int off = 16; off > 0; off >>= 1) v += __shfl_xor(v, off, 32);
  return v;
}

__device__ __forceinline__ float gelu_tanh(float x) {
  float x3 = x * x * x;
  return 0.5f * x * (1.0f + tanhf(0.7978845608028654f * (x + 0.044715f * x3)));
}

// ============================================================
// transpose fp32 (K x N) -> bf16 (N x K)
// ============================================================
__global__ __launch_bounds__(256) void transpose_f32_to_bf16(
    const float* __restrict__ in, __bf16* __restrict__ out, int K, int N) {
  __shared__ float tile[32][33];
  int n0 = blockIdx.x * 32;
  int k0 = blockIdx.y * 32;
  int c  = threadIdx.x & 31;
  int r4 = (threadIdx.x >> 5) * 4;
  #pragma unroll
  for (int i = 0; i < 4; i++)
    tile[r4 + i][c] = in[(size_t)(k0 + r4 + i) * N + n0 + c];
  __syncthreads();
  #pragma unroll
  for (int i = 0; i < 4; i++)
    out[(size_t)(n0 + r4 + i) * K + k0 + c] = (__bf16)tile[c][r4 + i];
}

// ============================================================
// mod = silu(vec) @ mod_w + mod_b        (2 x 9216)
// ============================================================
__global__ __launch_bounds__(256) void mod_kernel(
    const float* __restrict__ vec, const float* __restrict__ mod_w,
    const float* __restrict__ mod_b, float* __restrict__ mod) {
  __shared__ float sv[2 * HIDN];
  for (int i = threadIdx.x; i < 2 * HIDN; i += 256) {
    float x = vec[i];
    sv[i] = x / (1.0f + __expf(-x));
  }
  __syncthreads();
  int n = blockIdx.x * 256 + threadIdx.x;   // [0, 9216)
  float a0 = 0.f, a1 = 0.f;
  for (int k = 0; k < HIDN; k++) {
    float w = mod_w[(size_t)k * QKV_N + n];
    a0 += sv[k] * w;
    a1 += sv[HIDN + k] * w;
  }
  float b = mod_b[n];
  mod[n]         = a0 + b;
  mod[QKV_N + n] = a1 + b;
}

// ============================================================
// x_mod = (1+scale)*LN(x) + shift  -> bf16 (A matrix for GEMM1)
// ============================================================
__global__ __launch_bounds__(256) void ln_mod_kernel(
    const float* __restrict__ x, const float* __restrict__ mod,
    __bf16* __restrict__ x_mod) {
  __shared__ float r1[8], r2[8];
  int row  = blockIdx.x;           // token
  int b    = row >> 11;            // /2048
  int lane = threadIdx.x & 31, wave = threadIdx.x >> 5;
  const float* xr = x + (size_t)row * HIDN;
  float s = 0.f, s2 = 0.f;
  for (int i = threadIdx.x; i < HIDN; i += 256) {
    float v = xr[i]; s += v; s2 += v * v;
  }
  s = wave_sum(s); s2 = wave_sum(s2);
  if (lane == 0) { r1[wave] = s; r2[wave] = s2; }
  __syncthreads();
  if (threadIdx.x == 0) {
    float S = 0.f, S2 = 0.f;
    #pragma unroll
    for (int i = 0; i < 8; i++) { S += r1[i]; S2 += r2[i]; }
    r1[0] = S; r2[0] = S2;
  }
  __syncthreads();
  float mu   = r1[0] * (1.0f / HIDN);
  float var  = r2[0] * (1.0f / HIDN) - mu * mu;
  float rstd = rsqrtf(var + 1e-6f);
  const float* shiftp = mod + (size_t)b * QKV_N;
  const float* scalep = shiftp + HIDN;
  for (int i = threadIdx.x; i < HIDN; i += 256) {
    float v = (xr[i] - mu) * rstd;
    x_mod[(size_t)row * HIDN + i] = (__bf16)((1.0f + scalep[i]) * v + shiftp[i]);
  }
}

// ============================================================
// bf16 WMMA GEMM:  C(MxN) = A(MxK) * Bt(NxK)^T  [+ fused epilogue]
// block tile 128x128x32, 8 wave32 waves, wave tile 32x64 (2x4 WMMA)
// double-buffered LDS; async global->LDS DMA when available
// MODE 0: GEMM1 epilogue (bias, split qkv f32 / gelu->concat bf16)
// MODE 1: GEMM2 epilogue (bias, residual + gate)
// ============================================================
#define BM 128
#define BN 128
#define BK 32
#define LDSS 40   // bf16 stride (32 cols + 8 pad) -> 80B, keeps 16B alignment

template <int MODE>
__global__ __launch_bounds__(256) void gemm_bf16_wmma(
    const __bf16* __restrict__ A,   // M x K
    const __bf16* __restrict__ Bt,  // N x K (pre-transposed weights)
    int K, int N,
    const float* __restrict__ bias,     // N
    float*  __restrict__ out_qkv,       // MODE0: M x 9216 f32
    __bf16* __restrict__ out_cat,       // MODE0: M x 15360 bf16 (cols 3072..)
    const float* __restrict__ xres,     // MODE1: residual x
    const float* __restrict__ modv,     // MODE1: mod (2 x 9216), gate at +6144
    float* __restrict__ out_final) {    // MODE1: M x 3072 f32
#ifdef USE_ASYNC_LDS
  __shared__ __bf16 At[2][BM * LDSS];
  __shared__ __bf16 Bl[2][BN * LDSS];
#else
  __shared__ __bf16 At[1][BM * LDSS];
  __shared__ __bf16 Bl[1][BN * LDSS];
#endif

  int tid = threadIdx.x, lane = tid & 31, wave = tid >> 5;
  int waveM = wave >> 1;     // 0..3 -> 32-row band
  int waveN = wave & 1;      // 0..1 -> 64-col band
  int m0 = blockIdx.y * BM, n0 = blockIdx.x * BN;

  v8f acc[2][4];
  #pragma unroll
  for (int m = 0; m < 2; m++)
    #pragma unroll
    for (int n = 0; n < 4; n++)
      acc[m][n] = (v8f)(0.0f);

  int khalf = lane >> 4;
  int lrow  = lane & 15;

  // per-thread staging coordinates (two 16B chunks per tile per matrix)
  int lin0 = tid * 8;            // pass 0
  int lin1 = (256 + tid) * 8;    // pass 1
  int r0 = lin0 >> 5, c0 = lin0 & 31;
  int r1_ = lin1 >> 5, c1 = lin1 & 31;

#ifdef USE_ASYNC_LDS
  int nT = K / BK;
  // prologue: tile 0 -> buffer 0 (4 async DMA instructions per wave)
  async_cp16(A  + (size_t)(m0 + r0) * K + c0, &At[0][r0 * LDSS + c0]);
  async_cp16(A  + (size_t)(m0 + r1_) * K + c1, &At[0][r1_ * LDSS + c1]);
  async_cp16(Bt + (size_t)(n0 + r0) * K + c0, &Bl[0][r0 * LDSS + c0]);
  async_cp16(Bt + (size_t)(n0 + r1_) * K + c1, &Bl[0][r1_ * LDSS + c1]);

  for (int t = 0; t < nT; t++) {
    int buf = t & 1;
    if (t + 1 < nT) {
      int kn = (t + 1) * BK, bn = (t + 1) & 1;
      async_cp16(A  + (size_t)(m0 + r0) * K + kn + c0, &At[bn][r0 * LDSS + c0]);
      async_cp16(A  + (size_t)(m0 + r1_) * K + kn + c1, &At[bn][r1_ * LDSS + c1]);
      async_cp16(Bt + (size_t)(n0 + r0) * K + kn + c0, &Bl[bn][r0 * LDSS + c0]);
      async_cp16(Bt + (size_t)(n0 + r1_) * K + kn + c1, &Bl[bn][r1_ * LDSS + c1]);
      __builtin_amdgcn_s_wait_asynccnt(4);  // tile t landed; tile t+1 in flight
    } else {
      __builtin_amdgcn_s_wait_asynccnt(0);
    }
    __syncthreads();

    union Frag { v16bf v; u32x4 u[2]; };
    Frag af[2], bf_[4];
    #pragma unroll
    for (int m = 0; m < 2; m++) {
      const __bf16* p = At[buf] + (waveM * 32 + m * 16 + lrow) * LDSS + khalf * 8;
      af[m].u[0] = *(const u32x4*)(p);
      af[m].u[1] = *(const u32x4*)(p + 16);
    }
    #pragma unroll
    for (int n = 0; n < 4; n++) {
      const __bf16* p = Bl[buf] + (waveN * 64 + n * 16 + lrow) * LDSS + khalf * 8;
      bf_[n].u[0] = *(const u32x4*)(p);
      bf_[n].u[1] = *(const u32x4*)(p + 16);
    }
    #pragma unroll
    for (int m = 0; m < 2; m++)
      #pragma unroll
      for (int n = 0; n < 4; n++)
        acc[m][n] = __builtin_amdgcn_wmma_f32_16x16x32_bf16(
            false, af[m].v, false, bf_[n].v, (short)0, acc[m][n], false, false);
    __syncthreads();
  }
#else
  for (int k0 = 0; k0 < K; k0 += BK) {
    {
      u32x4 da = *(const u32x4*)(A  + (size_t)(m0 + r0) * K + k0 + c0);
      *(u32x4*)(&At[0][r0 * LDSS + c0]) = da;
      u32x4 db = *(const u32x4*)(Bt + (size_t)(n0 + r0) * K + k0 + c0);
      *(u32x4*)(&Bl[0][r0 * LDSS + c0]) = db;
      u32x4 dc = *(const u32x4*)(A  + (size_t)(m0 + r1_) * K + k0 + c1);
      *(u32x4*)(&At[0][r1_ * LDSS + c1]) = dc;
      u32x4 dd = *(const u32x4*)(Bt + (size_t)(n0 + r1_) * K + k0 + c1);
      *(u32x4*)(&Bl[0][r1_ * LDSS + c1]) = dd;
    }
    __syncthreads();

    union Frag { v16bf v; u32x4 u[2]; };
    Frag af[2], bf_[4];
    #pragma unroll
    for (int m = 0; m < 2; m++) {
      const __bf16* p = At[0] + (waveM * 32 + m * 16 + lrow) * LDSS + khalf * 8;
      af[m].u[0] = *(const u32x4*)(p);
      af[m].u[1] = *(const u32x4*)(p + 16);
    }
    #pragma unroll
    for (int n = 0; n < 4; n++) {
      const __bf16* p = Bl[0] + (waveN * 64 + n * 16 + lrow) * LDSS + khalf * 8;
      bf_[n].u[0] = *(const u32x4*)(p);
      bf_[n].u[1] = *(const u32x4*)(p + 16);
    }
    #pragma unroll
    for (int m = 0; m < 2; m++)
      #pragma unroll
      for (int n = 0; n < 4; n++)
        acc[m][n] = __builtin_amdgcn_wmma_f32_16x16x32_bf16(
            false, af[m].v, false, bf_[n].v, (short)0, acc[m][n], false, false);
    __syncthreads();
  }
#endif

  // ---- epilogue (C/D layout: lane n = lane&15, M = (lane>>4)*8 + vgpr) ----
  #pragma unroll
  for (int m = 0; m < 2; m++) {
    #pragma unroll
    for (int n = 0; n < 4; n++) {
      int gn    = n0 + waveN * 64 + n * 16 + (lane & 15);
      int mbase = m0 + waveM * 32 + m * 16 + (lane >> 4) * 8;
      float bs  = bias[gn];
      #pragma unroll
      for (int v = 0; v < 8; v++) {
        int gm = mbase + v;
        float val = acc[m][n][v] + bs;
        if (MODE == 0) {
          if (gn < QKV_N) {
            out_qkv[(size_t)gm * QKV_N + gn] = val;
          } else {
            out_cat[(size_t)gm * K2 + HIDN + (gn - QKV_N)] = (__bf16)gelu_tanh(val);
          }
        } else {
          int bidx   = gm >> 11;  // /2048
          float gate = modv[(size_t)bidx * QKV_N + 2 * HIDN + gn];
          out_final[(size_t)gm * HIDN + gn] =
              xres[(size_t)gm * HIDN + gn] + gate * val;
        }
      }
    }
  }
}

// ============================================================
// per-token: RMS-norm(q,k) + RoPE + 24x24 head-mix softmax
// writes attn (bf16) into concat cols [0,3072)
// ============================================================
__global__ __launch_bounds__(256) void attn_kernel(
    const float* __restrict__ qkv, const float* __restrict__ pe,
    const float* __restrict__ q_scale, const float* __restrict__ k_scale,
    __bf16* __restrict__ out_cat) {
  __shared__ float qs[NHEADS][HD];
  __shared__ float ks[NHEADS][HD];
  __shared__ float vs[NHEADS][HD];
  __shared__ float sc[NHEADS][NHEADS + 1];

  int row = blockIdx.x;                       // token = b*L + l
  const float* base = qkv + (size_t)row * QKV_N;
  const float* pe_t = pe + (size_t)row * 256; // (64 pairs) x 2 x 2
  int tid = threadIdx.x, lane = tid & 31, wave = tid >> 5;

  // copy V
  for (int i = tid; i < HIDN; i += 256) vs[i >> 7][i & 127] = base[2 * HIDN + i];

  // 48 vectors (24 q + 24 k): one wave each, rms-norm + rope
  for (int vid = wave; vid < 48; vid += 8) {
    int h = vid % NHEADS, isK = vid / NHEADS;
    const float* src = base + isK * HIDN + h * HD;
    const float* scl = isK ? k_scale : q_scale;
    float4 xv = *(const float4*)(src + lane * 4);
    float ss = xv.x * xv.x + xv.y * xv.y + xv.z * xv.z + xv.w * xv.w;
    ss = wave_sum(ss);
    float rr = rsqrtf(ss * (1.0f / HD) + 1e-6f);
    float xn0 = xv.x * rr * scl[4 * lane + 0];
    float xn1 = xv.y * rr * scl[4 * lane + 1];
    float xn2 = xv.z * rr * scl[4 * lane + 2];
    float xn3 = xv.w * rr * scl[4 * lane + 3];
    const float* f0 = pe_t + (2 * lane) * 4;      // pair 2*lane
    const float* f1 = pe_t + (2 * lane + 1) * 4;  // pair 2*lane+1
    float* dst = (isK ? ks[h] : qs[h]) + 4 * lane;
    dst[0] = f0[0] * xn0 + f0[1] * xn1;
    dst[1] = f0[2] * xn0 + f0[3] * xn1;
    dst[2] = f1[0] * xn2 + f1[1] * xn3;
    dst[3] = f1[2] * xn2 + f1[3] * xn3;
  }
  __syncthreads();

  // scores (24x24), scale 1/sqrt(128)
  for (int s = tid; s < NHEADS * NHEADS; s += 256) {
    int h = s / NHEADS, g = s % NHEADS;
    float d = 0.f;
    #pragma unroll 8
    for (int e = 0; e < HD; e++) d += qs[h][e] * ks[g][e];
    sc[h][g] = d * 0.08838834764831845f;
  }
  __syncthreads();

  // softmax over g per head
  if (tid < NHEADS) {
    float mx = -1e30f;
    #pragma unroll
    for (int g = 0; g < NHEADS; g++) mx = fmaxf(mx, sc[tid][g]);
    float se = 0.f;
    #pragma unroll
    for (int g = 0; g < NHEADS; g++) { float e = __expf(sc[tid][g] - mx); sc[tid][g] = e; se += e; }
    float inv = 1.0f / se;
    #pragma unroll
    for (int g = 0; g < NHEADS; g++) sc[tid][g] *= inv;
  }
  __syncthreads();

  // attn[h][d] = sum_g p[h][g] * v[g][d]  -> concat cols [0,3072)
  for (int o = tid; o < HIDN; o += 256) {
    int h = o >> 7, d = o & 127;
    float a = 0.f;
    #pragma unroll
    for (int g = 0; g < NHEADS; g++) a += sc[h][g] * vs[g][d];
    out_cat[(size_t)row * K2 + o] = (__bf16)a;
  }
}

// ============================================================
// launch
// ============================================================
extern "C" void kernel_launch(void* const* d_in, const int* in_sizes, int n_in,
                              void* d_out, int out_size, void* d_ws, size_t ws_size,
                              hipStream_t stream) {
  const float* x       = (const float*)d_in[0];
  const float* vec     = (const float*)d_in[1];
  const float* pe      = (const float*)d_in[2];
  const float* mod_w   = (const float*)d_in[3];
  const float* mod_b   = (const float*)d_in[4];
  const float* lin1_w  = (const float*)d_in[5];
  const float* lin1_b  = (const float*)d_in[6];
  const float* lin2_w  = (const float*)d_in[7];
  const float* lin2_b  = (const float*)d_in[8];
  const float* q_scale = (const float*)d_in[9];
  const float* k_scale = (const float*)d_in[10];
  float* out = (float*)d_out;

  char* ws = (char*)d_ws;
  float*  mod    = (float*)(ws);                                  // 2*9216 f32
  __bf16* x_modb = (__bf16*)(ws + 73728);                         // 4096*3072 bf16
  __bf16* w1t    = (__bf16*)(ws + 73728 + 25165824ull);           // 21504*3072 bf16
  __bf16* w2t    = (__bf16*)(ws + 73728 + 25165824ull + 132120576ull);   // 3072*15360
  float*  qkv    = (float*)(ws + 73728 + 25165824ull + 132120576ull + 94371840ull);
  __bf16* catb   = (__bf16*)(ws + 73728 + 25165824ull + 132120576ull + 94371840ull
                                + 150994944ull);                  // 4096*15360 bf16

  // weight convert + transpose (fp32 KxN -> bf16 NxK)
  transpose_f32_to_bf16<<<dim3(N1 / 32, K1 / 32), 256, 0, stream>>>(lin1_w, w1t, K1, N1);
  transpose_f32_to_bf16<<<dim3(N2 / 32, K2 / 32), 256, 0, stream>>>(lin2_w, w2t, K2, N2);

  // modulation vector
  mod_kernel<<<QKV_N / 256, 256, 0, stream>>>(vec, mod_w, mod_b, mod);

  // layernorm + (1+scale)/shift -> bf16 A matrix
  ln_mod_kernel<<<M_TOT, 256, 0, stream>>>(x, mod, x_modb);

  // GEMM1: h = x_mod @ lin1_w + b1 ; split qkv (f32) / gelu(mlp) -> concat
  gemm_bf16_wmma<0><<<dim3(N1 / BN, M_TOT / BM), 256, 0, stream>>>(
      x_modb, w1t, K1, N1, lin1_b, qkv, catb, nullptr, nullptr, nullptr);

  // per-token head-mix attention -> concat cols [0,3072)
  attn_kernel<<<M_TOT, 256, 0, stream>>>(qkv, pe, q_scale, k_scale, catb);

  // GEMM2: out = x + gate * (concat @ lin2_w + b2)
  gemm_bf16_wmma<1><<<dim3(N2 / BN, M_TOT / BM), 256, 0, stream>>>(
      catb, w2t, K2, N2, lin2_b, nullptr, nullptr, x, mod, out);
}
